// LIGAAttention_2327872274510
// MI455X (gfx1250) — matrix operand order repaired and checked by reading, not compiled, not run on hardware
//
#include <hip/hip_runtime.h>
#include <hip/hip_bf16.h>
#include <cstdint>
#include <cstddef>

typedef __bf16 bf16_t;
typedef __attribute__((ext_vector_type(8)))  __bf16 v8bf;
typedef __attribute__((ext_vector_type(16))) __bf16 v16bf;
typedef __attribute__((ext_vector_type(8)))  float  v8f;

static constexpr int BATCH = 2;
static constexpr int VLEN  = 2048;
static constexpr int TLEN  = 448;
static constexpr int KLEN  = 64;
static constexpr int TOT   = VLEN + TLEN + KLEN;   // 2560
static constexpr int MIDL  = VLEN + TLEN;          // 2496
static constexpr int HDIM  = 1024;
static constexpr int NHEAD = 16;
static constexpr int DHEAD = 64;

// ---------------------------------------------------------------------------
// WMMA helpers (CDNA5 16x16x32 bf16, f32 accumulate)
// ---------------------------------------------------------------------------
__device__ __forceinline__ v8f wmma_bf(v16bf a, v16bf b, v8f c) {
  return __builtin_amdgcn_wmma_f32_16x16x32_bf16(
      /*neg_a=*/false, a, /*neg_b=*/false, b,
      /*c_mod=*/(short)0, c, /*reuse_a=*/false, /*reuse_b=*/false);
}

// A fragment: 16 rows x 32 K (bf16), row-major source with leading dim ld.
// ISA layout: lanes 0-15 -> row=lane, K {0..7,16..23}; lanes 16-31 -> K {8..15,24..31}.
__device__ __forceinline__ v16bf load_a_frag(const bf16_t* src, int ld) {
  const int lane = threadIdx.x & 31;
  const int row  = lane & 15;
  const int hi   = lane >> 4;
  const bf16_t* p = src + (size_t)row * ld + hi * 8;
  v8bf lo  = *(const v8bf*)(p);
  v8bf hi8 = *(const v8bf*)(p + 16);
  v16bf f;
#pragma unroll
  for (int i = 0; i < 8; ++i) { f[i] = lo[i]; f[i + 8] = hi8[i]; }
  return f;
}

// B fragment: source stored as N rows x K cols (i.e. B^T row-major), leading dim ld.
// ISA layout: lane holds col n = lane&15; lanes<16 -> K {0..7,16..23}; lanes>=16 -> K {8..15,24..31}.
__device__ __forceinline__ v16bf load_b_frag(const bf16_t* src, int ld) {
  const int lane = threadIdx.x & 31;
  const bf16_t* p = src + (size_t)(lane & 15) * ld + (lane >> 4) * 8;
  v8bf lo  = *(const v8bf*)(p);
  v8bf hi8 = *(const v8bf*)(p + 16);
  v16bf f;
#pragma unroll
  for (int i = 0; i < 8; ++i) { f[i] = lo[i]; f[i + 8] = hi8[i]; }
  return f;
}

// ---------------------------------------------------------------------------
// Elementwise kernels
// ---------------------------------------------------------------------------
__global__ void cast_f32_bf16(const float* __restrict__ s, bf16_t* __restrict__ d, int n) {
  int i = blockIdx.x * 256 + threadIdx.x;
  if (i < n) d[i] = (bf16_t)s[i];
}

__global__ void concat_cast(const float* __restrict__ v, const float* __restrict__ t,
                            const float* __restrict__ k, bf16_t* __restrict__ hs) {
  int i = blockIdx.x * 256 + threadIdx.x;
  if (i >= BATCH * TOT * HDIM) return;
  int c   = i % HDIM;
  int tok = (i / HDIM) % TOT;
  int b   = i / (TOT * HDIM);
  float val;
  if (tok < VLEN)             val = v[((size_t)b * VLEN + tok) * HDIM + c];
  else if (tok < VLEN + TLEN) val = t[((size_t)b * TLEN + (tok - VLEN)) * HDIM + c];
  else                        val = k[((size_t)b * KLEN + (tok - VLEN - TLEN)) * HDIM + c];
  hs[i] = (bf16_t)val;
}

// v0 [B][tok][h*64+d] -> v0t [B][h][d][tok]
__global__ void transpose_v_kernel(const bf16_t* __restrict__ v0, bf16_t* __restrict__ v0t) {
  int i = blockIdx.x * 256 + threadIdx.x;
  if (i >= BATCH * TOT * HDIM) return;
  int d   = i & 63;
  int h   = (i >> 6) & 15;
  int tok = (i >> 10) % TOT;
  int b   = i / (TOT * HDIM);
  v0t[(((size_t)b * NHEAD + h) * DHEAD + d) * TOT + tok] = v0[i];
}

// Copy task slices into q1/k1/v1t (note: kv = q0 task slice per reference).
__global__ void seed_task(const bf16_t* __restrict__ q0, const bf16_t* __restrict__ k0,
                          bf16_t* __restrict__ q1, bf16_t* __restrict__ k1,
                          bf16_t* __restrict__ v1t) {
  int i = blockIdx.x * 256 + threadIdx.x;
  if (i >= BATCH * KLEN * HDIM) return;
  int d = i & 63;
  int h = (i >> 6) & 15;
  int t = (i >> 10) & 63;
  int b = i >> 16;
  size_t src = (((size_t)b * TOT + MIDL + t) * NHEAD + h) * DHEAD + d;
  size_t dst = (((size_t)b * TOT + t) * NHEAD + h) * DHEAD + d;
  q1[dst] = q0[src];
  k1[dst] = k0[src];
  v1t[(((size_t)b * NHEAD + h) * DHEAD + d) * TOT + t] = q0[src];  // kv = q0 slice
}

// ---------------------------------------------------------------------------
// Register-blocked WMMA GEMM: C[M,N] = A[M,K] * W[N,K]^T + bias.
// 4 waves/block; each wave owns a 32x64 tile (2 M-subtiles x 4 N-subtiles,
// 8 accumulators). Per k=32 step: 12 b128 loads for 8 wmmas (1.5 loads/wmma),
// A fragments reused 4x, B fragments 2x from registers.
// Grid: (M/32, N/256).
// ---------------------------------------------------------------------------
__global__ void gemm_bf16(const bf16_t* __restrict__ A, int lda,
                          const bf16_t* __restrict__ W, int ldb,
                          const float* __restrict__ bias,
                          void* __restrict__ C, int ldc, int K, int out_f32) {
  const int lane = threadIdx.x & 31;
  const int wv   = threadIdx.x >> 5;
  const int mt   = blockIdx.x;                  // 32-row tile index
  const int nt0  = (blockIdx.y * 4 + wv) * 4;   // first of 4 16-col subtiles
  const bf16_t* a = A + (size_t)mt * 32 * lda;
  const bf16_t* w = W + (size_t)nt0 * 16 * ldb;

  v8f acc[2][4] = {};
  for (int k = 0; k < K; k += 32) {
    v16bf a0 = load_a_frag(a + k, lda);
    v16bf a1 = load_a_frag(a + (size_t)16 * lda + k, lda);
#pragma unroll
    for (int j = 0; j < 4; ++j) {
      v16bf bf = load_b_frag(w + (size_t)j * 16 * ldb + k, ldb);
      acc[0][j] = wmma_bf(a0, bf, acc[0][j]);
      acc[1][j] = wmma_bf(a1, bf, acc[1][j]);
    }
  }

  const int hi = lane >> 4, col = lane & 15;
#pragma unroll
  for (int j = 0; j < 4; ++j) {
    const int n  = (nt0 + j) * 16 + col;
    const float bv = bias ? bias[n] : 0.0f;
#pragma unroll
    for (int i = 0; i < 2; ++i)
#pragma unroll
      for (int r = 0; r < 8; ++r) {
        size_t row = (size_t)mt * 32 + i * 16 + 8 * hi + r;
        float val  = acc[i][j][r] + bv;
        if (out_f32) ((float*)C)[row * ldc + n]  = val;
        else         ((bf16_t*)C)[row * ldc + n] = (bf16_t)val;
      }
  }
}

// ---------------------------------------------------------------------------
// Flash attention: one wave per (16-query tile, head, batch).
// q/k: [B][btoks][NHEAD][DHEAD] bf16;  vt: [B][NHEAD][DHEAD][vt_ld] (V^T).
// scores scaled by `scale` (reference uses 1/D), streaming softmax over Lk.
// ---------------------------------------------------------------------------
__global__ void attn_flash(const bf16_t* __restrict__ q, int q_btoks, int q_tok0,
                           const bf16_t* __restrict__ kk, int k_btoks, int k_tok0, int Lk,
                           const bf16_t* __restrict__ vt, int vt_ld, int vt_col0,
                           bf16_t* __restrict__ out, int o_btoks, int o_tok0,
                           float scale) {
  __shared__ __align__(32) bf16_t ldsp[16 * 32];
  const int lane = threadIdx.x & 31;
  const int qt = blockIdx.x, h = blockIdx.y, b = blockIdx.z;
  const int ts = NHEAD * DHEAD;  // token stride in halfs

  const bf16_t* qb = q + (((size_t)b * q_btoks + q_tok0 + qt * 16) * NHEAD + h) * DHEAD;
  v16bf qa0 = load_a_frag(qb, ts);
  v16bf qa1 = load_a_frag(qb + 32, ts);

  v8f acc[4] = {};
  float mrun[8], lrun[8];
#pragma unroll
  for (int r = 0; r < 8; ++r) { mrun[r] = -1e30f; lrun[r] = 0.0f; }

  const int hi = lane >> 4, col = lane & 15;
  const bf16_t* vbase = vt + (((size_t)b * NHEAD + h) * DHEAD) * (size_t)vt_ld + vt_col0;

  for (int kt = 0; kt < Lk; kt += 32) {
    const bf16_t* kb = kk + (((size_t)b * k_btoks + k_tok0 + kt) * NHEAD + h) * DHEAD;
    v16bf k00 = load_b_frag(kb, ts);
    v16bf k01 = load_b_frag(kb + 32, ts);
    v16bf k10 = load_b_frag(kb + (size_t)16 * ts, ts);
    v16bf k11 = load_b_frag(kb + (size_t)16 * ts + 32, ts);
    v8f s0 = {}; s0 = wmma_bf(qa0, k00, s0); s0 = wmma_bf(qa1, k01, s0);
    v8f s1 = {}; s1 = wmma_bf(qa0, k10, s1); s1 = wmma_bf(qa1, k11, s1);

    float p0[8], p1[8], m8[8];
#pragma unroll
    for (int r = 0; r < 8; ++r) {
      p0[r] = s0[r] * scale; p1[r] = s1[r] * scale;
      m8[r] = fmaxf(p0[r], p1[r]);
    }
#pragma unroll
    for (int m = 1; m < 16; m <<= 1)
#pragma unroll
      for (int r = 0; r < 8; ++r) m8[r] = fmaxf(m8[r], __shfl_xor(m8[r], m, 32));

    float fac[8];
#pragma unroll
    for (int r = 0; r < 8; ++r) {
      float mn = fmaxf(mrun[r], m8[r]);
      fac[r] = __expf(mrun[r] - mn);
      mrun[r] = mn;
      p0[r] = __expf(p0[r] - mn);
      p1[r] = __expf(p1[r] - mn);
    }
    float rs[8];
#pragma unroll
    for (int r = 0; r < 8; ++r) rs[r] = p0[r] + p1[r];
#pragma unroll
    for (int m = 1; m < 16; m <<= 1)
#pragma unroll
      for (int r = 0; r < 8; ++r) rs[r] += __shfl_xor(rs[r], m, 32);
#pragma unroll
    for (int r = 0; r < 8; ++r) lrun[r] = lrun[r] * fac[r] + rs[r];
#pragma unroll
    for (int t = 0; t < 4; ++t)
#pragma unroll
      for (int r = 0; r < 8; ++r) acc[t][r] *= fac[r];

    // probs (C/D layout) -> LDS -> A fragment
#pragma unroll
    for (int r = 0; r < 8; ++r) {
      int row = r + 8 * hi;
      ldsp[row * 32 + col]      = (bf16_t)p0[r];
      ldsp[row * 32 + 16 + col] = (bf16_t)p1[r];
    }
    asm volatile("s_wait_dscnt 0" ::: "memory");
    __builtin_amdgcn_wave_barrier();
    v16bf pa = load_a_frag(ldsp, 32);
    asm volatile("s_wait_dscnt 0" ::: "memory");
    __builtin_amdgcn_wave_barrier();

#pragma unroll
    for (int t = 0; t < 4; ++t) {
      v16bf vf = load_b_frag(vbase + (size_t)(t * 16) * vt_ld + kt, vt_ld);
      acc[t] = wmma_bf(pa, vf, acc[t]);
    }
  }

  float inv[8];
#pragma unroll
  for (int r = 0; r < 8; ++r) inv[r] = 1.0f / lrun[r];
  bf16_t* ob = out + (((size_t)b * o_btoks + o_tok0 + qt * 16) * NHEAD + h) * DHEAD;
#pragma unroll
  for (int t = 0; t < 4; ++t)
#pragma unroll
    for (int r = 0; r < 8; ++r)
      ob[(size_t)(r + 8 * hi) * ts + t * 16 + col] = (bf16_t)(acc[t][r] * inv[r]);
}

// ---------------------------------------------------------------------------
// Per-head 64x64 linears on mid -> q1/k1 (normal) and v1 (transposed).
// One wave per 16 rows of the flattened [B*MIDL*NHEAD, 64] A matrix.
// ---------------------------------------------------------------------------
__global__ void midlin_kernel(const bf16_t* __restrict__ mid,
                              const bf16_t* __restrict__ wq, const bf16_t* __restrict__ wk,
                              const bf16_t* __restrict__ wv,
                              const float* __restrict__ bq, const float* __restrict__ bk,
                              const float* __restrict__ bv,
                              bf16_t* __restrict__ q1, bf16_t* __restrict__ k1,
                              bf16_t* __restrict__ v1t) {
  const int lane = threadIdx.x & 31;
  const int tile = blockIdx.x;
  const bf16_t* a = mid + (size_t)tile * 16 * DHEAD;
  v16bf a0 = load_a_frag(a, DHEAD);
  v16bf a1 = load_a_frag(a + 32, DHEAD);
  const int hi = lane >> 4, col = lane & 15;
#pragma unroll
  for (int nt = 0; nt < 4; ++nt) {
    v8f cq = {}, ck = {}, cv = {};
    cq = wmma_bf(a0, load_b_frag(wq + nt * 16 * DHEAD, DHEAD), cq);
    cq = wmma_bf(a1, load_b_frag(wq + nt * 16 * DHEAD + 32, DHEAD), cq);
    ck = wmma_bf(a0, load_b_frag(wk + nt * 16 * DHEAD, DHEAD), ck);
    ck = wmma_bf(a1, load_b_frag(wk + nt * 16 * DHEAD + 32, DHEAD), ck);
    cv = wmma_bf(a0, load_b_frag(wv + nt * 16 * DHEAD, DHEAD), cv);
    cv = wmma_bf(a1, load_b_frag(wv + nt * 16 * DHEAD + 32, DHEAD), cv);
    const int n = nt * 16 + col;
    const float bqv = bq[n], bkv = bk[n], bvv = bv[n];
#pragma unroll
    for (int r = 0; r < 8; ++r) {
      int g = tile * 16 + 8 * hi + r;
      int h = g & 15;
      int t = (g >> 4) % MIDL;
      int b = g / (16 * MIDL);
      size_t o = (((size_t)b * TOT + KLEN + t) * NHEAD + h) * DHEAD + n;
      q1[o] = (bf16_t)(cq[r] + bqv);
      k1[o] = (bf16_t)(ck[r] + bkv);
      v1t[(((size_t)b * NHEAD + h) * DHEAD + n) * TOT + KLEN + t] = (bf16_t)(cv[r] + bvv);
    }
  }
}

// ---------------------------------------------------------------------------
extern "C" void kernel_launch(void* const* d_in, const int* in_sizes, int n_in,
                              void* d_out, int out_size, void* d_ws, size_t ws_size,
                              hipStream_t stream) {
  (void)in_sizes; (void)n_in; (void)out_size; (void)ws_size;
  const float* vision = (const float*)d_in[0];
  const float* text   = (const float*)d_in[1];
  const float* task   = (const float*)d_in[2];
  const float* wq0 = (const float*)d_in[3];  const float* bq0 = (const float*)d_in[4];
  const float* wk0 = (const float*)d_in[5];  const float* bk0 = (const float*)d_in[6];
  const float* wv0 = (const float*)d_in[7];  const float* bv0 = (const float*)d_in[8];
  const float* wq1 = (const float*)d_in[9];  const float* bq1 = (const float*)d_in[10];
  const float* wk1 = (const float*)d_in[11]; const float* bk1 = (const float*)d_in[12];
  const float* wv1 = (const float*)d_in[13]; const float* bv1 = (const float*)d_in[14];
  const float* wo  = (const float*)d_in[15]; const float* bo  = (const float*)d_in[16];

  uint8_t* base = (uint8_t*)d_ws;
  size_t off = 0;
  auto carve = [&](size_t elems) -> bf16_t* {
    bf16_t* r = (bf16_t*)(base + off);
    off = (off + elems * sizeof(bf16_t) + 255) & ~(size_t)255;
    return r;
  };
  const size_t NTH = (size_t)BATCH * TOT * HDIM;  // 5,242,880
  bf16_t* hs   = carve(NTH);
  bf16_t* wq0b = carve(HDIM * HDIM);
  bf16_t* wk0b = carve(HDIM * HDIM);
  bf16_t* wv0b = carve(HDIM * HDIM);
  bf16_t* wob  = carve(HDIM * HDIM);
  bf16_t* wq1b = carve(DHEAD * DHEAD);
  bf16_t* wk1b = carve(DHEAD * DHEAD);
  bf16_t* wv1b = carve(DHEAD * DHEAD);
  bf16_t* q0   = carve(NTH);
  bf16_t* k0   = carve(NTH);
  bf16_t* v0   = carve(NTH);
  bf16_t* v0t  = carve(NTH);
  bf16_t* mid  = carve((size_t)BATCH * MIDL * HDIM);
  bf16_t* q1   = carve(NTH);
  bf16_t* k1   = carve(NTH);
  bf16_t* v1t  = carve(NTH);
  bf16_t* att2 = carve(NTH);

  // 1) casts
  concat_cast<<<(int)((NTH + 255) / 256), 256, 0, stream>>>(vision, text, task, hs);
  cast_f32_bf16<<<(HDIM * HDIM + 255) / 256, 256, 0, stream>>>(wq0, wq0b, HDIM * HDIM);
  cast_f32_bf16<<<(HDIM * HDIM + 255) / 256, 256, 0, stream>>>(wk0, wk0b, HDIM * HDIM);
  cast_f32_bf16<<<(HDIM * HDIM + 255) / 256, 256, 0, stream>>>(wv0, wv0b, HDIM * HDIM);
  cast_f32_bf16<<<(HDIM * HDIM + 255) / 256, 256, 0, stream>>>(wo,  wob,  HDIM * HDIM);
  cast_f32_bf16<<<(DHEAD * DHEAD + 255) / 256, 256, 0, stream>>>(wq1, wq1b, DHEAD * DHEAD);
  cast_f32_bf16<<<(DHEAD * DHEAD + 255) / 256, 256, 0, stream>>>(wk1, wk1b, DHEAD * DHEAD);
  cast_f32_bf16<<<(DHEAD * DHEAD + 255) / 256, 256, 0, stream>>>(wv1, wv1b, DHEAD * DHEAD);

  // 2) QKV projections (M = B*TOT = 5120, N = K = 1024), 32x64 per wave
  dim3 gP(BATCH * TOT / 32, HDIM / 256);
  gemm_bf16<<<gP, 128, 0, stream>>>(hs, HDIM, wq0b, HDIM, bq0, q0, HDIM, HDIM, 0);
  gemm_bf16<<<gP, 128, 0, stream>>>(hs, HDIM, wk0b, HDIM, bk0, k0, HDIM, HDIM, 0);
  gemm_bf16<<<gP, 128, 0, stream>>>(hs, HDIM, wv0b, HDIM, bv0, v0, HDIM, HDIM, 0);

  // 3) V^T for PV fragments
  transpose_v_kernel<<<(int)((NTH + 255) / 256), 256, 0, stream>>>(v0, v0t);

  const float scale = 1.0f / (float)DHEAD;  // reference divides by d, not sqrt(d)
  // 4) cross attentions -> mid (v2t rows 0..2047, t2v rows 2048..2495)
  attn_flash<<<dim3(VLEN / 16, NHEAD, BATCH), 32, 0, stream>>>(
      q0, TOT, 0,            k0, TOT, VLEN, TLEN,  v0t, TOT, VLEN,
      mid, MIDL, 0, scale);
  attn_flash<<<dim3(TLEN / 16, NHEAD, BATCH), 32, 0, stream>>>(
      q0, TOT, VLEN,         k0, TOT, 0, VLEN,     v0t, TOT, 0,
      mid, MIDL, VLEN, scale);

  // 5) per-head linears -> q1/k1/v1t (+ task-slice seed)
  midlin_kernel<<<BATCH * MIDL * NHEAD / 16, 32, 0, stream>>>(
      mid, wq1b, wk1b, wv1b, bq1, bk1, bv1, q1, k1, v1t);
  seed_task<<<(BATCH * KLEN * HDIM + 255) / 256, 256, 0, stream>>>(q0, k0, q1, k1, v1t);

  // 6) full attention over all 2560 tokens
  attn_flash<<<dim3(TOT / 16, NHEAD, BATCH), 32, 0, stream>>>(
      q1, TOT, 0,  k1, TOT, 0, TOT,  v1t, TOT, 0,
      att2, TOT, 0, scale);

  // 7) output projection (fp32 out)
  gemm_bf16<<<gP, 128, 0, stream>>>(att2, HDIM, wob, HDIM, bo, d_out, HDIM, HDIM, 1);
}
